// PointNetLocal_44753559224571
// MI455X (gfx1250) — matrix operand order repaired and checked by reading, not compiled, not run on hardware
//
#include <hip/hip_runtime.h>
#include <math.h>

typedef __attribute__((ext_vector_type(16))) _Float16 v16h;
typedef __attribute__((ext_vector_type(8)))  _Float16 v8h;
typedef __attribute__((ext_vector_type(2)))  _Float16 v2h;
typedef __attribute__((ext_vector_type(8)))  float    v8f;

#define B_DIM       8
#define N_PTS       100000
#define C_DIM       256
#define OUT_DIM     16
#define CELLS_PAD   4913          // 17*17*17
#define TILES_PER_B 6250          // 100000 / 16
#define TOTAL_TILES (B_DIM * TILES_PER_B)
#define MAIN_BLOCKS 512

__device__ __forceinline__ float neg_inf_f() {
    union { unsigned u; float f; } c; c.u = 0xff800000u; return c.f;
}

// ---------------------------------------------------------------------------
// Pass 1: fill the whole padded output with -inf (atomic-max identity).
// ---------------------------------------------------------------------------
__global__ void pnl_init_kernel(float* __restrict__ out, int n) {
    int i = blockIdx.x * blockDim.x + threadIdx.x;
    if (i < n) out[i] = neg_inf_f();
}

// ---------------------------------------------------------------------------
// Pass 3: -inf -> 0 (covers empty cells AND the zero padding region, which
// atomics never touch).
// ---------------------------------------------------------------------------
__global__ void pnl_finalize_kernel(float* __restrict__ out, int n) {
    int i = blockIdx.x * blockDim.x + threadIdx.x;
    if (i < n) {
        float v = out[i];
        union { float f; unsigned u; } c; c.f = v;
        out[i] = (c.u == 0xff800000u) ? 0.0f : v;
    }
}

// ---------------------------------------------------------------------------
// Pass 2: per-wave tile pipeline.
//   layer1 (K=3) on VALU -> ReLU -> f16 into per-wave LDS slab
//   layer2 (K=256) as 8x v_wmma_f32_16x16x32_f16
//   scatter-max of D into padded grid via float atomicMax
// ---------------------------------------------------------------------------
__launch_bounds__(256)
__global__ void pnl_main_kernel(const float* __restrict__ x,
                                const float* __restrict__ W1,
                                const float* __restrict__ b1,
                                const float* __restrict__ W2,
                                const float* __restrict__ b2,
                                float* __restrict__ out) {
    __shared__ float    w1s[C_DIM * 3];          // 3 KB
    __shared__ float    b1s[C_DIM];              // 1 KB
    __shared__ _Float16 hbuf[8][16 * C_DIM];     // 8 waves x 8 KB = 64 KB
    __shared__ int      cells[8][16];

    const int tid = threadIdx.x;

    // Stage W1 / b1 once per workgroup (broadcast reads afterwards).
    for (int i = tid; i < C_DIM * 3; i += 256) w1s[i] = W1[i];
    for (int i = tid; i < C_DIM; i += 256)     b1s[i] = b1[i];
    __syncthreads();

    const int lane = tid & 31;
    const int slot = tid >> 5;        // wave slot within workgroup
    const int n    = lane & 15;       // output channel (WMMA N) / layer-1 point id
    const int half = lane >> 4;       // 0 or 1
    const int kbA  = half * 8;        // A-fragment K base (f16 16x32 layout)
    const int kbB  = half * 8;        // B-fragment K base (f16 32x16 layout)
    const int c0   = half * 128;      // layer-1 channel range base for this lane
    const int mb   = half * 8;        // C/D fragment M base

    // Build resident B fragments: bfrag[q] covers K = 32q..32q+31 of W2^T.
    // Lane holds column n, rows kbB..kbB+7 and kbB+16..kbB+23 of W2 row n.
    v16h bfrag[8];
#pragma unroll
    for (int q = 0; q < 8; ++q) {
        const float* src = W2 + (size_t)n * C_DIM + q * 32 + kbB;
#pragma unroll
        for (int j = 0; j < 16; ++j) bfrag[q][j] = (_Float16)src[j < 8 ? j : j + 8];
    }
    const float bias2 = b2[n];

    const int wave_gid = blockIdx.x * 8 + slot;
    const int nwaves   = gridDim.x * 8;

    for (int tile = wave_gid; tile < TOTAL_TILES; tile += nwaves) {
        const int b  = tile / TILES_PER_B;
        const int p0 = (tile - b * TILES_PER_B) * 16;

        // Coordinates of this lane's layer-1 point (lanes L and L+16 share a
        // point; they split the 256 channels).
        const size_t base = ((size_t)b * N_PTS + p0 + n) * 3;
        const float x0 = x[base + 0];
        const float x1 = x[base + 1];
        const float x2 = x[base + 2];

        if (lane < 16) {
            int ix = min(max((int)floorf(x0), 0), 15);
            int iy = min(max((int)floorf(x1), 0), 15);
            int iz = min(max((int)floorf(x2), 0), 15);
            cells[slot][lane] = ix * 289 + iy * 17 + iz;   // padded-grid offset
        }

        // Layer 1: 128 channels per lane, packed f16 pair stores into LDS.
        _Float16* hrow = &hbuf[slot][n * C_DIM];
#pragma unroll 4
        for (int c = c0; c < c0 + 128; c += 2) {
            float v0 = fmaf(x2, w1s[c * 3 + 2],
                       fmaf(x1, w1s[c * 3 + 1],
                       fmaf(x0, w1s[c * 3 + 0], b1s[c])));
            float v1 = fmaf(x2, w1s[c * 3 + 5],
                       fmaf(x1, w1s[c * 3 + 4],
                       fmaf(x0, w1s[c * 3 + 3], b1s[c + 1])));
            v0 = fmaxf(v0, 0.0f);
            v1 = fmaxf(v1, 0.0f);
            v2h pk;
            pk.x = (_Float16)v0;
            pk.y = (_Float16)v1;
            *(v2h*)(hrow + c) = pk;
        }

        // Cross-lane LDS produce->consume within the wave: drain DS counter.
        asm volatile("s_wait_dscnt 0" ::: "memory");

        // Layer 2: D = h(16x256) x W2^T(256x16) + b2, eight K=32 WMMA steps.
        v8f acc;
#pragma unroll
        for (int r = 0; r < 8; ++r) acc[r] = bias2;

        const v8h* hp = (const v8h*)&hbuf[slot][0];
#pragma unroll
        for (int q = 0; q < 8; ++q) {
            const int e  = n * C_DIM + q * 32 + kbA;       // f16 element index
            v8h a_lo = hp[e >> 3];                         // K kbA .. kbA+7
            v8h a_hi = hp[(e + 16) >> 3];                  // K kbA+16 .. kbA+23
            v16h a = __builtin_shufflevector(a_lo, a_hi,
                        0, 1, 2, 3, 4, 5, 6, 7, 8, 9, 10, 11, 12, 13, 14, 15);
            acc = __builtin_amdgcn_wmma_f32_16x16x32_f16(
                      false, a, false, bfrag[q], (short)0, acc, false, false);
        }

        // Scatter-max: lane owns D(m = mb + r, n) for r = 0..7.
        const size_t obase = ((size_t)b * OUT_DIM + n) * CELLS_PAD;
#pragma unroll
        for (int r = 0; r < 8; ++r) {
            const int cell = cells[slot][mb + r];
            atomicMax(&out[obase + cell], acc[r]);
        }
    }
}

extern "C" void kernel_launch(void* const* d_in, const int* in_sizes, int n_in,
                              void* d_out, int out_size, void* d_ws, size_t ws_size,
                              hipStream_t stream) {
    const float* x  = (const float*)d_in[0];
    const float* W1 = (const float*)d_in[1];
    const float* b1 = (const float*)d_in[2];
    const float* W2 = (const float*)d_in[3];
    const float* b2 = (const float*)d_in[4];
    float* out = (float*)d_out;

    const int n = out_size;  // 8 * 16 * 17 * 17 * 17 = 628864
    pnl_init_kernel<<<(n + 255) / 256, 256, 0, stream>>>(out, n);
    pnl_main_kernel<<<MAIN_BLOCKS, 256, 0, stream>>>(x, W1, b1, W2, b2, out);
    pnl_finalize_kernel<<<(n + 255) / 256, 256, 0, stream>>>(out, n);
}
